// Transformer_76003741270239
// MI455X (gfx1250) — matrix operand order repaired and checked by reading, not compiled, not run on hardware
//
#include <hip/hip_runtime.h>
#include <cstdint>

typedef __attribute__((ext_vector_type(16))) _Float16 v16h;
typedef __attribute__((ext_vector_type(8)))  _Float16 v8h;
typedef __attribute__((ext_vector_type(8)))  float    v8f;
typedef __attribute__((ext_vector_type(4)))  int      v4i;

#define DEPTH 6
#define DIM   192
#define HEADS 12
#define DH    16
#define MLP   768
#define NTOK  256
#define BATCH 32
#define INNER 192
#define ROWS  (BATCH * NTOK)   // 8192 tokens

// ---------------------------------------------------------------------------
// gfx1250 async copy (GLOBAL_LOAD_ASYNC_TO_LDS_B128, ASYNCcnt-tracked)
// Builtin signature (from hipcc diagnostic): (v4i AS1*, v4i AS3*, Ii, Ii)
// ---------------------------------------------------------------------------
#if defined(__AMDGCN__) && defined(__has_builtin)
#if __has_builtin(__builtin_amdgcn_global_load_async_to_lds_b128)
#define HAVE_ASYNC_COPY 1
#endif
#endif
#ifndef HAVE_ASYNC_COPY
#define HAVE_ASYNC_COPY 0
#endif

#if HAVE_ASYNC_COPY
typedef __attribute__((address_space(1))) v4i* gv4p;
typedef __attribute__((address_space(3))) v4i* lv4p;

static __device__ __forceinline__ void async_copy16(const void* g, uint32_t lds_off) {
  __builtin_amdgcn_global_load_async_to_lds_b128(
      (gv4p)(uintptr_t)g, (lv4p)(uintptr_t)lds_off, 0, 0);
}
static __device__ __forceinline__ void async_wait0() {
#if __has_builtin(__builtin_amdgcn_s_wait_asynccnt)
  __builtin_amdgcn_s_wait_asynccnt(0);
#else
  asm volatile("s_wait_asynccnt 0" ::: "memory");
#endif
}
#endif

// ---------------------------------------------------------------------------
// WMMA helpers (CDNA5 ISA 7.12.2 layouts, wave32)
// ---------------------------------------------------------------------------
static __device__ __forceinline__ v8f wmma16x16x32(v16h a, v16h b, v8f c) {
  // (neg_a, A, neg_b, B, c_mod, C, reuse_a, reuse_b)
  return __builtin_amdgcn_wmma_f32_16x16x32_f16(false, a, false, b, (short)0, c,
                                                false, false);
}

// A fragment: 16(M) x 32(K), row-major source.
// Per-lane row = lane&15, K = (e&7) + 8*(lane>=16) + 16*(e>=8):
// each half-fragment is 16 contiguous bytes -> two 128-bit loads.
static __device__ __forceinline__ v16h ld_a_frag(const _Float16* src, int stride) {
  int l = threadIdx.x & 31;
  const _Float16* p = src + (l & 15) * stride + ((l >> 4) << 3);
  v8h lo = *(const v8h*)p;
  v8h hi = *(const v8h*)(p + 16);
  v16h f;
#pragma unroll
  for (int e = 0; e < 8; ++e) { f[e] = lo[e]; f[e + 8] = hi[e]; }
  return f;
}

// A fragment with only K<16 valid (DH=16 padded to 32 with zeros)
static __device__ __forceinline__ v16h ld_a_frag_k16(const _Float16* src, int stride) {
  int l = threadIdx.x & 31;
  int row = l & 15;
  int kb = (l >> 4) << 3;
  v16h f;
#pragma unroll
  for (int e = 0; e < 16; ++e) {
    int k = (e & 7) + kb + ((e >> 3) << 4);
    f[e] = (k < DH) ? src[row * stride + k] : (_Float16)0.0f;
  }
  return f;
}

// B fragment: 32(K) x 16(N). src is B^T row-major (N x K): src[n*stride + k].
// Per-lane col = lane&15, K = (e&7) + 8*(lane>=16) + 16*(e>=8) -> two 16-byte loads.
static __device__ __forceinline__ v16h ld_b_frag(const _Float16* src, int stride) {
  int l = threadIdx.x & 31;
  const _Float16* p = src + (l & 15) * stride + ((l >> 4) << 3);
  v8h lo = *(const v8h*)p;
  v8h hi = *(const v8h*)(p + 16);
  v16h f;
#pragma unroll
  for (int e = 0; e < 8; ++e) { f[e] = lo[e]; f[e + 8] = hi[e]; }
  return f;
}

static __device__ __forceinline__ v16h ld_b_frag_k16(const _Float16* src, int stride) {
  int l = threadIdx.x & 31;
  int n = l & 15;
  int kb = (l >> 4) << 3;
  v16h f;
#pragma unroll
  for (int e = 0; e < 16; ++e) {
    int k = (e & 7) + kb + ((e >> 3) << 4);
    f[e] = (k < DH) ? src[n * stride + k] : (_Float16)0.0f;
  }
  return f;
}

// ---------------------------------------------------------------------------
// Weight convert + transpose: Wt[l][n][k] = (f16) W[l][k][n]
// ---------------------------------------------------------------------------
__global__ void k_wcvt(const float* __restrict__ W, _Float16* __restrict__ Wt,
                       int K, int N) {
  long per = (long)K * N;
  long total = (long)DEPTH * per;
  for (long i = (long)blockIdx.x * blockDim.x + threadIdx.x; i < total;
       i += (long)gridDim.x * blockDim.x) {
    long l = i / per;
    long r = i - l * per;
    int n = (int)(r / K), k = (int)(r % K);
    Wt[i] = (_Float16)W[l * per + (long)k * N + n];
  }
}

// ---------------------------------------------------------------------------
// LayerNorm: one wave per token row (DIM=192 -> 6 f32 per lane)
// ---------------------------------------------------------------------------
__global__ __launch_bounds__(32) void k_ln(const float* __restrict__ x,
                                           const float* __restrict__ g,
                                           const float* __restrict__ bta,
                                           _Float16* __restrict__ h) {
  int row = blockIdx.x;
  int l = threadIdx.x;
  const float* xr = x + (long)row * DIM;
  float v[6];
  float s = 0.f;
#pragma unroll
  for (int t = 0; t < 6; ++t) { v[t] = xr[l + 32 * t]; s += v[t]; }
#pragma unroll
  for (int o = 16; o > 0; o >>= 1) s += __shfl_xor(s, o, 32);
  float mean = s * (1.0f / DIM);
  float q = 0.f;
#pragma unroll
  for (int t = 0; t < 6; ++t) { float d = v[t] - mean; q += d * d; }
#pragma unroll
  for (int o = 16; o > 0; o >>= 1) q += __shfl_xor(q, o, 32);
  float inv = rsqrtf(q * (1.0f / DIM) + 1e-5f);
  _Float16* hr = h + (long)row * DIM;
#pragma unroll
  for (int t = 0; t < 6; ++t) {
    int c = l + 32 * t;
    hr[c] = (_Float16)((v[t] - mean) * inv * g[c] + bta[c]);
  }
}

// ---------------------------------------------------------------------------
// Generic WMMA GEMM: C[M,N] = A[M,K] @ B[K,N], A f16 (lda==K), B as f16 B^T.
// Block = 256 threads = 8 waves -> 32x64 output tile; A strip staged in LDS
// via GLOBAL_LOAD_ASYNC_TO_LDS_B128 when available.
// ---------------------------------------------------------------------------
constexpr int EPI_STORE = 0;  // store f16
constexpr int EPI_GELU  = 1;  // +bias, exact GELU, store f16
constexpr int EPI_RES   = 2;  // (+bias)*ls added into f32 residual

template <int EPI, int KD>
__global__ __launch_bounds__(256) void k_gemm(const _Float16* __restrict__ A,
                                              const _Float16* __restrict__ Bt,
                                              const float* __restrict__ bias,
                                              const float* __restrict__ ls,
                                              float* __restrict__ xres,
                                              _Float16* __restrict__ outh,
                                              int Ncols) {
  __shared__ alignas(16) _Float16 sA[32 * KD];
  int mblk = blockIdx.x;
  int nblk = blockIdx.y;

  const _Float16* Ab = A + (long)mblk * 32 * KD;
#if HAVE_ASYNC_COPY
  {
    uint32_t sbase = (uint32_t)(uintptr_t)sA;
    const char* gbase = (const char*)Ab;
    for (int i = threadIdx.x; i < (32 * KD * 2) / 16; i += 256)
      async_copy16(gbase + (size_t)i * 16, sbase + (uint32_t)i * 16);
    async_wait0();
  }
#else
  for (int i = threadIdx.x; i < (32 * KD) / 2; i += 256)
    ((uint32_t*)sA)[i] = ((const uint32_t*)Ab)[i];
#endif
  __syncthreads();

  int w = threadIdx.x >> 5;
  int wm = w & 1, wn = w >> 1;
  int row0 = wm * 16;              // row within LDS strip
  int col0 = nblk * 64 + wn * 16;  // global output column
  const _Float16* Bp = Bt + (long)col0 * KD;

  v8f acc = {};
#pragma unroll
  for (int kk = 0; kk < KD; kk += 32) {
    __builtin_prefetch(Bp + kk + 64, 0, 1);  // global_prefetch_b8
    v16h a = ld_a_frag(sA + row0 * KD + kk, KD);
    v16h b = ld_b_frag(Bp + kk, KD);
    acc = wmma16x16x32(a, b, acc);
  }

  int lane = threadIdx.x & 31;
  int n_ = lane & 15;
  int gm0 = mblk * 32 + row0;
#pragma unroll
  for (int r = 0; r < 8; ++r) {
    int m_ = r + ((lane >> 4) << 3);
    int gm = gm0 + m_;
    int gn = col0 + n_;
    float c = acc[r];
    if constexpr (EPI == EPI_STORE) {
      outh[(long)gm * Ncols + gn] = (_Float16)c;
    } else if constexpr (EPI == EPI_GELU) {
      c += bias[gn];
      c = 0.5f * c * (1.0f + erff(c * 0.7071067811865476f));
      outh[(long)gm * Ncols + gn] = (_Float16)c;
    } else {
      c = (c + bias[gn]) * ls[gn];
      xres[(long)gm * Ncols + gn] += c;
    }
  }
}

// ---------------------------------------------------------------------------
// Fused talking-heads LSA attention.
// Grid: (BATCH, NTOK/16). Block: 12 waves (one per head), 384 threads.
// Dynamic LDS: mixed scores [HEADS][16][NTOK] f32 + raw tile [HEADS][16][16].
// ---------------------------------------------------------------------------
#define ATTN_SMEM ((HEADS * 16 * NTOK + HEADS * 16 * 16) * (int)sizeof(float))

__global__ __launch_bounds__(384) void k_attn(const _Float16* __restrict__ q,
                                              const _Float16* __restrict__ kv,
                                              const float* __restrict__ scale,
                                              const float* __restrict__ pre,
                                              const float* __restrict__ post,
                                              _Float16* __restrict__ o) {
  extern __shared__ float smem[];
  float* sDots = smem;                      // [HEADS][16][NTOK]
  float* sRaw = smem + HEADS * 16 * NTOK;   // [HEADS][16][16]

  int b = blockIdx.x;
  int i0 = blockIdx.y * 16;
  int head = threadIdx.x >> 5;   // wave id == head id
  int lane = threadIdx.x & 31;
  int n_ = lane & 15;

  // --- scores: q @ k^T (K = DH = 16, padded to 32) + scale + diag + mix_pre ---
  const _Float16* qb = q + ((long)b * NTOK + i0) * INNER + head * DH;
  v16h qa = ld_a_frag_k16(qb, INNER);
  float sc = scale[head];
  float prew[HEADS];
#pragma unroll
  for (int h = 0; h < HEADS; ++h) prew[h] = pre[h * HEADS + head];

  const _Float16* kbase = kv + (long)b * NTOK * (2 * INNER) + head * DH;
  for (int jb = 0; jb < 16; ++jb) {
    int j0 = jb * 16;
    v16h kf = ld_b_frag_k16(kbase + (long)j0 * (2 * INNER), 2 * INNER);
    v8f d = {};
    d = wmma16x16x32(qa, kf, d);
#pragma unroll
    for (int r = 0; r < 8; ++r) {
      int m_ = r + ((lane >> 4) << 3);
      float val = d[r] * sc;
      if (i0 + m_ == j0 + n_) val = -1e-9f;   // LSA diagonal
      sRaw[(head * 16 + m_) * 16 + n_] = val;
    }
    __syncthreads();
    // talking-heads pre-mix: out head g=head, sum over input heads h
#pragma unroll
    for (int r = 0; r < 8; ++r) {
      int m_ = r + ((lane >> 4) << 3);
      float acc2 = 0.f;
#pragma unroll
      for (int h = 0; h < HEADS; ++h)
        acc2 += sRaw[(h * 16 + m_) * 16 + n_] * prew[h];
      sDots[(head * 16 + m_) * NTOK + j0 + n_] = acc2;
    }
    __syncthreads();
  }

  // --- softmax over 256 keys, per (head, row); wave-local ---
  for (int r = 0; r < 16; ++r) {
    float* rowp = sDots + (head * 16 + r) * NTOK;
    float mx = -1e30f;
#pragma unroll
    for (int t = 0; t < 8; ++t) mx = fmaxf(mx, rowp[lane + 32 * t]);
#pragma unroll
    for (int off = 16; off > 0; off >>= 1) mx = fmaxf(mx, __shfl_xor(mx, off, 32));
    float e[8];
    float sum = 0.f;
#pragma unroll
    for (int t = 0; t < 8; ++t) { e[t] = __expf(rowp[lane + 32 * t] - mx); sum += e[t]; }
#pragma unroll
    for (int off = 16; off > 0; off >>= 1) sum += __shfl_xor(sum, off, 32);
    float inv = 1.0f / sum;
#pragma unroll
    for (int t = 0; t < 8; ++t) rowp[lane + 32 * t] = e[t] * inv;
  }
  __syncthreads();

  // --- post-mix (on the fly) + A @ V via WMMA, output head = head ---
  float postw[HEADS];
#pragma unroll
  for (int g = 0; g < HEADS; ++g) postw[g] = post[g * HEADS + head];
  const _Float16* vb = kv + (long)b * NTOK * (2 * INNER) + INNER + head * DH;

  int row_ = lane & 15;
  int kb8 = (lane >> 4) << 3;
  int kb16 = (lane >> 4) << 4;
  v8f oacc = {};
  for (int jc = 0; jc < NTOK; jc += 32) {
    v16h af;
#pragma unroll
    for (int e = 0; e < 16; ++e) {
      int j = jc + (e & 7) + kb8 + ((e >> 3) << 4);
      float s2 = 0.f;
#pragma unroll
      for (int g = 0; g < HEADS; ++g)
        s2 += sDots[(g * 16 + row_) * NTOK + j] * postw[g];
      af[e] = (_Float16)s2;
    }
    v16h vf;
#pragma unroll
    for (int e = 0; e < 16; ++e) {
      int j = jc + (e & 7) + kb8 + ((e >> 3) << 4);
      vf[e] = vb[(long)j * (2 * INNER) + n_];
    }
    oacc = wmma16x16x32(af, vf, oacc);
  }
  _Float16* ob = o + ((long)b * NTOK + i0) * INNER + head * DH;
#pragma unroll
  for (int r = 0; r < 8; ++r) {
    int m_ = r + ((lane >> 4) << 3);
    ob[(long)m_ * INNER + n_] = (_Float16)oacc[r];
  }
}

// ---------------------------------------------------------------------------
// Host orchestration
// ---------------------------------------------------------------------------
static inline void* ws_alloc(void* base, size_t& off, size_t bytes) {
  void* p = (char*)base + off;
  off += (bytes + 255) & ~(size_t)255;
  return p;
}

extern "C" void kernel_launch(void* const* d_in, const int* in_sizes, int n_in,
                              void* d_out, int out_size, void* d_ws, size_t ws_size,
                              hipStream_t stream) {
  (void)in_sizes; (void)n_in; (void)out_size; (void)ws_size;

  const float* in_x    = (const float*)d_in[0];
  const float* ln1_g   = (const float*)d_in[1];
  const float* ln1_b   = (const float*)d_in[2];
  const float* Wq      = (const float*)d_in[3];
  const float* Wkv     = (const float*)d_in[4];
  const float* lsa     = (const float*)d_in[5];
  const float* mix_pre = (const float*)d_in[6];
  const float* mix_post= (const float*)d_in[7];
  const float* Wo      = (const float*)d_in[8];
  const float* bo      = (const float*)d_in[9];
  const float* ls1     = (const float*)d_in[10];
  const float* ln2_g   = (const float*)d_in[11];
  const float* ln2_b   = (const float*)d_in[12];
  const float* W1      = (const float*)d_in[13];
  const float* b1      = (const float*)d_in[14];
  const float* W2      = (const float*)d_in[15];
  const float* b2      = (const float*)d_in[16];
  const float* ls2     = (const float*)d_in[17];

  float* x = (float*)d_out;  // x updated in place across layers

  size_t off = 0;
  _Float16* wtq  = (_Float16*)ws_alloc(d_ws, off, (size_t)DEPTH * DIM * INNER * 2);
  _Float16* wtkv = (_Float16*)ws_alloc(d_ws, off, (size_t)DEPTH * DIM * 2 * INNER * 2);
  _Float16* wto  = (_Float16*)ws_alloc(d_ws, off, (size_t)DEPTH * INNER * DIM * 2);
  _Float16* wt1  = (_Float16*)ws_alloc(d_ws, off, (size_t)DEPTH * DIM * MLP * 2);
  _Float16* wt2  = (_Float16*)ws_alloc(d_ws, off, (size_t)DEPTH * MLP * DIM * 2);
  _Float16* hbuf = (_Float16*)ws_alloc(d_ws, off, (size_t)ROWS * DIM * 2);
  _Float16* qbuf = (_Float16*)ws_alloc(d_ws, off, (size_t)ROWS * INNER * 2);
  _Float16* kvb  = (_Float16*)ws_alloc(d_ws, off, (size_t)ROWS * 2 * INNER * 2);
  _Float16* obuf = (_Float16*)ws_alloc(d_ws, off, (size_t)ROWS * INNER * 2);
  _Float16* f1b  = (_Float16*)ws_alloc(d_ws, off, (size_t)ROWS * MLP * 2);

  (void)hipMemcpyAsync(x, in_x, (size_t)ROWS * DIM * sizeof(float),
                       hipMemcpyDeviceToDevice, stream);

  // One-time (per call) weight convert + transpose to f16 N x K layout
  k_wcvt<<<1024, 256, 0, stream>>>(Wq,  wtq,  DIM,   INNER);
  k_wcvt<<<1024, 256, 0, stream>>>(Wkv, wtkv, DIM,   2 * INNER);
  k_wcvt<<<1024, 256, 0, stream>>>(Wo,  wto,  INNER, DIM);
  k_wcvt<<<1024, 256, 0, stream>>>(W1,  wt1,  DIM,   MLP);
  k_wcvt<<<1024, 256, 0, stream>>>(W2,  wt2,  MLP,   DIM);

  for (int l = 0; l < DEPTH; ++l) {
    // PreNorm 1
    k_ln<<<ROWS, 32, 0, stream>>>(x, ln1_g + l * DIM, ln1_b + l * DIM, hbuf);
    // Q and KV projections
    k_gemm<EPI_STORE, DIM><<<dim3(ROWS / 32, INNER / 64), 256, 0, stream>>>(
        hbuf, wtq + (long)l * DIM * INNER, nullptr, nullptr, nullptr, qbuf, INNER);
    k_gemm<EPI_STORE, DIM><<<dim3(ROWS / 32, (2 * INNER) / 64), 256, 0, stream>>>(
        hbuf, wtkv + (long)l * DIM * 2 * INNER, nullptr, nullptr, nullptr, kvb,
        2 * INNER);
    // Fused talking-heads attention
    k_attn<<<dim3(BATCH, NTOK / 16), 384, ATTN_SMEM, stream>>>(
        qbuf, kvb, lsa + l * HEADS, mix_pre + l * HEADS * HEADS,
        mix_post + l * HEADS * HEADS, obuf);
    // Output projection + LayerScale + residual
    k_gemm<EPI_RES, DIM><<<dim3(ROWS / 32, DIM / 64), 256, 0, stream>>>(
        obuf, wto + (long)l * INNER * DIM, bo + l * DIM, ls1 + l * DIM, x,
        nullptr, DIM);
    // PreNorm 2 + MLP
    k_ln<<<ROWS, 32, 0, stream>>>(x, ln2_g + l * DIM, ln2_b + l * DIM, hbuf);
    k_gemm<EPI_GELU, DIM><<<dim3(ROWS / 32, MLP / 64), 256, 0, stream>>>(
        hbuf, wt1 + (long)l * DIM * MLP, b1 + l * MLP, nullptr, nullptr, f1b, MLP);
    k_gemm<EPI_RES, MLP><<<dim3(ROWS / 32, DIM / 64), 256, 0, stream>>>(
        f1b, wt2 + (long)l * MLP * DIM, b2 + l * DIM, ls2 + l * DIM, x,
        nullptr, DIM);
  }
}
